// GB_GLHF_1288490189083
// MI455X (gfx1250) — matrix operand order repaired and checked by reading, not compiled, not run on hardware
//
#include <hip/hip_runtime.h>
#include <hip/hip_bf16.h>
#include <math.h>

typedef __attribute__((ext_vector_type(16))) _Float16 v16h;
typedef __attribute__((ext_vector_type(8)))  _Float16 v8h;
typedef __attribute__((ext_vector_type(8)))  float    v8f;

static constexpr int Bb  = 8;
static constexpr int Nn  = 1000;
static constexpr int Dd  = 2000;
static constexpr int H2  = 100;
static constexpr int QK  = 1000;
static constexpr int NP2 = 1024;   // N padded to 32-row / 64-col tiles
static constexpr int DP  = 2048;   // D padded to 64-col tiles
static constexpr int KP  = 1024;   // K padded to 32
static constexpr int D1  = Dd + 1; // 2001

// ---------------------------------------------------------------- sort -----
__global__ void sort_kernel(const float* __restrict__ bp, int* __restrict__ order) {
  __shared__ float key[1024];
  __shared__ int   idx[1024];
  const int b = blockIdx.x, t = threadIdx.x;
  key[t] = (t < Nn) ? bp[((size_t)b * Nn + t) * D1] : __builtin_inff();
  idx[t] = t;
  __syncthreads();
  for (int k = 2; k <= 1024; k <<= 1) {
    for (int j = k >> 1; j > 0; j >>= 1) {
      int ixj = t ^ j;
      if (ixj > t) {
        bool up = ((t & k) == 0);
        float a = key[t], c = key[ixj];
        int ia = idx[t], ic = idx[ixj];
        bool gt = (a > c) || (a == c && ia > ic);
        if (gt == up) { key[t] = c; key[ixj] = a; idx[t] = ic; idx[ixj] = ia; }
      }
      __syncthreads();
    }
  }
  if (t < Nn) order[b * Nn + t] = idx[t];
}

// ------------------------------------------------ fitness stats + ftoken ----
__global__ void ftoken_kernel(const float* __restrict__ bp, const int* __restrict__ order,
                              float* __restrict__ fit_sorted, float* __restrict__ ftoken) {
  __shared__ float s1[256], s2[256];
  const int b = blockIdx.x, t = threadIdx.x;
  float sum = 0.f, sq = 0.f;
  for (int i = t; i < Nn; i += 256) {
    float f = bp[((size_t)b * Nn + order[b * Nn + i]) * D1];
    fit_sorted[b * Nn + i] = f;
    sum += f; sq += f * f;
  }
  s1[t] = sum; s2[t] = sq; __syncthreads();
  for (int o = 128; o > 0; o >>= 1) {
    if (t < o) { s1[t] += s1[t + o]; s2[t] += s2[t + o]; }
    __syncthreads();
  }
  float mean = s1[0] / (float)Nn;
  float var  = (s2[0] - (float)Nn * mean * mean) / (float)(Nn - 1);
  float istd = 1.f / sqrtf(var);
  const float rmean = (float)(Nn - 1) * 0.5f;
  const float rstd  = sqrtf(((float)Nn * ((float)Nn * (float)Nn - 1.f) / 12.f) / (float)(Nn - 1));
  for (int i = t; i < Nn; i += 256) {
    float f = fit_sorted[b * Nn + i];
    ftoken[(size_t)(b * Nn + i) * 2 + 0] = (f - mean) * istd;
    ftoken[(size_t)(b * Nn + i) * 2 + 1] = ((float)i - rmean) / rstd;
  }
}

// ---------------------------------------------------------- h = LN(relu) ----
__global__ void h_kernel(const float* __restrict__ ftoken,
                         const float* __restrict__ w_w, const float* __restrict__ w_b,
                         const float* __restrict__ g,   const float* __restrict__ be,
                         float* __restrict__ h) {
  int token = blockIdx.x * blockDim.x + threadIdx.x;
  if (token >= Bb * Nn) return;
  float f = ftoken[(size_t)token * 2], r = ftoken[(size_t)token * 2 + 1];
  float sum = 0.f, sq = 0.f;
  for (int o = 0; o < H2; ++o) {
    float a = fmaxf(f * w_w[o * 2] + r * w_w[o * 2 + 1] + w_b[o], 0.f);
    sum += a; sq += a * a;
  }
  float mu  = sum / (float)H2;
  float inv = rsqrtf(sq / (float)H2 - mu * mu + 1e-5f);
  for (int o = 0; o < H2; ++o) {
    float a = fmaxf(f * w_w[o * 2] + r * w_w[o * 2 + 1] + w_b[o], 0.f);
    h[(size_t)token * H2 + o] = (a - mu) * inv * g[o] + be[o];
  }
}

// -------------------------------------------- q1/k1 (f16, zero-padded) -----
__global__ void qk_kernel(const float* __restrict__ h,
                          const float* __restrict__ fq_w, const float* __restrict__ fq_b,
                          const float* __restrict__ fk_w, const float* __restrict__ fk_b,
                          _Float16* __restrict__ q1, _Float16* __restrict__ k1) {
  __shared__ float hs[H2];
  const int bn = blockIdx.x;           // b*NP2 + n
  const int b = bn >> 10, n = bn & (NP2 - 1);
  const int t = threadIdx.x;
  const bool valid = (n < Nn);
  if (valid && t < H2) hs[t] = h[((size_t)b * Nn + n) * H2 + t];
  __syncthreads();
  const size_t base = ((size_t)b * NP2 + n) * KP;
  for (int k = t; k < KP; k += 256) {
    float q = 0.f, kk = 0.f;
    if (valid && k < QK) {
      float aq = fq_b[k], ak = fk_b[k];
      const float* wq = fq_w + (size_t)k * H2;
      const float* wk = fk_w + (size_t)k * H2;
      for (int j = 0; j < H2; ++j) { aq += hs[j] * wq[j]; ak += hs[j] * wk[j]; }
      q = tanhf(aq); kk = tanhf(ak);
    }
    q1[base + k] = (_Float16)q;
    k1[base + k] = (_Float16)kk;
  }
}

// ------------------------------------------ chrom^T gather (f16, padded) ----
__global__ void chromT_kernel(const float* __restrict__ bp, const int* __restrict__ order,
                              _Float16* __restrict__ chromT) {
  size_t i = (size_t)blockIdx.x * 256 + threadIdx.x;
  const size_t total = (size_t)Bb * DP * KP;
  if (i >= total) return;
  int m = (int)(i & (KP - 1));
  size_t rest = i >> 10;
  int d = (int)(rest & (DP - 1));
  int b = (int)(rest >> 11);
  float val = 0.f;
  if (d < Dd && m < Nn) val = bp[((size_t)b * Nn + order[b * Nn + m]) * D1 + 1 + d];
  chromT[i] = (_Float16)val;
}

// ------------------------------------------------- WMMA fragment loaders ----
__device__ __forceinline__ v16h load_a_frag(const _Float16* row, int k0, int half) {
  union { v16h v; v8h h[2]; } a;
  a.h[0] = *(const v8h*)(row + k0 + half * 8);
  a.h[1] = *(const v8h*)(row + k0 + half * 8 + 16);
  return a.v;
}
__device__ __forceinline__ v16h load_b_frag(const _Float16* row, int k0, int half) {
  union { v16h v; v8h h[2]; } bf;
  bf.h[0] = *(const v8h*)(row + k0 + half * 8);
  bf.h[1] = *(const v8h*)(row + k0 + half * 8 + 16);
  return bf.v;
}
__device__ __forceinline__ v8f wmma_step(v16h a, v16h b, v8f c) {
  return __builtin_amdgcn_wmma_f32_16x16x32_f16(false, a, false, b, (short)0, c, false, false);
}

// ---------------------------------------------------------------- GEMM1 ----
// A = tanh(q1 . k1^T / sqrt(QK)) * mask ; 32x64 tile per wave
__global__ void gemm1_kernel(const _Float16* __restrict__ q1, const _Float16* __restrict__ k1,
                             const float* __restrict__ mut_rand,
                             float* __restrict__ Aout, _Float16* __restrict__ Ah) {
  const int b = blockIdx.z;
  const int rowBase = blockIdx.x * 32;
  const int colBase = blockIdx.y * 64;
  const int lane = threadIdx.x;
  const int half = lane >> 4;
  const int l15  = lane & 15;
  const _Float16* ar0 = q1 + ((size_t)b * NP2 + rowBase + l15) * KP;
  const _Float16* ar1 = ar0 + (size_t)16 * KP;
  const _Float16* br0 = k1 + ((size_t)b * NP2 + colBase + l15) * KP;
  const _Float16* br1 = br0 + (size_t)16 * KP;
  const _Float16* br2 = br0 + (size_t)32 * KP;
  const _Float16* br3 = br0 + (size_t)48 * KP;
  v8f acc[8];
  for (int i = 0; i < 8; ++i) acc[i] = (v8f){};
  for (int k0 = 0; k0 < KP; k0 += 32) {
    v16h a0 = load_a_frag(ar0, k0, half);
    v16h a1 = load_a_frag(ar1, k0, half);
    v16h b0 = load_b_frag(br0, k0, half);
    v16h b1 = load_b_frag(br1, k0, half);
    v16h b2 = load_b_frag(br2, k0, half);
    v16h b3 = load_b_frag(br3, k0, half);
    acc[0] = wmma_step(a0, b0, acc[0]);
    acc[1] = wmma_step(a0, b1, acc[1]);
    acc[2] = wmma_step(a0, b2, acc[2]);
    acc[3] = wmma_step(a0, b3, acc[3]);
    acc[4] = wmma_step(a1, b0, acc[4]);
    acc[5] = wmma_step(a1, b1, acc[5]);
    acc[6] = wmma_step(a1, b2, acc[6]);
    acc[7] = wmma_step(a1, b3, acc[7]);
  }
  const float scale = 0.031622776601683794f; // 1/sqrt(1000)
  for (int i = 0; i < 2; ++i) {
    for (int j = 0; j < 4; ++j) {
      v8f c = acc[i * 4 + j];
      int cc = colBase + j * 16 + l15;
      for (int v = 0; v < 8; ++v) {
        int rr = rowBase + i * 16 + half * 8 + v;
        float outv = 0.f;
        if (rr < Nn && cc < Nn) {
          float val = tanhf(c[v] * scale);
          float mm = (mut_rand[((size_t)b * Nn + rr) * Nn + cc] >= 0.5f) ? 1.f : 0.f;
          if (rr == cc) mm = 1.f;
          outv = val * mm;
          Aout[((size_t)b * Nn + rr) * Nn + cc] = outv;
        }
        Ah[((size_t)b * NP2 + rr) * KP + cc] = (_Float16)outv;
      }
    }
  }
}

// ---------------------------------------------------------------- GEMM2 ----
// vchrom = A . chrom   (A: f16 padded, chrom^T: f16 padded) ; 32x64 tile
__global__ void gemm2_kernel(const _Float16* __restrict__ Ah, const _Float16* __restrict__ chromT,
                             float* __restrict__ vchrom) {
  const int b = blockIdx.z;
  const int rowBase = blockIdx.x * 32;   // n
  const int colBase = blockIdx.y * 64;   // d
  const int lane = threadIdx.x;
  const int half = lane >> 4;
  const int l15  = lane & 15;
  const _Float16* ar0 = Ah + ((size_t)b * NP2 + rowBase + l15) * KP;
  const _Float16* ar1 = ar0 + (size_t)16 * KP;
  const _Float16* br0 = chromT + ((size_t)b * DP + colBase + l15) * KP;
  const _Float16* br1 = br0 + (size_t)16 * KP;
  const _Float16* br2 = br0 + (size_t)32 * KP;
  const _Float16* br3 = br0 + (size_t)48 * KP;
  v8f acc[8];
  for (int i = 0; i < 8; ++i) acc[i] = (v8f){};
  for (int k0 = 0; k0 < KP; k0 += 32) {
    v16h a0 = load_a_frag(ar0, k0, half);
    v16h a1 = load_a_frag(ar1, k0, half);
    v16h b0 = load_b_frag(br0, k0, half);
    v16h b1 = load_b_frag(br1, k0, half);
    v16h b2 = load_b_frag(br2, k0, half);
    v16h b3 = load_b_frag(br3, k0, half);
    acc[0] = wmma_step(a0, b0, acc[0]);
    acc[1] = wmma_step(a0, b1, acc[1]);
    acc[2] = wmma_step(a0, b2, acc[2]);
    acc[3] = wmma_step(a0, b3, acc[3]);
    acc[4] = wmma_step(a1, b0, acc[4]);
    acc[5] = wmma_step(a1, b1, acc[5]);
    acc[6] = wmma_step(a1, b2, acc[6]);
    acc[7] = wmma_step(a1, b3, acc[7]);
  }
  for (int i = 0; i < 2; ++i) {
    for (int j = 0; j < 4; ++j) {
      v8f c = acc[i * 4 + j];
      int cc = colBase + j * 16 + l15;
      for (int v = 0; v < 8; ++v) {
        int rr = rowBase + i * 16 + half * 8 + v;
        if (rr < Nn && cc < Dd) vchrom[((size_t)b * Nn + rr) * Dd + cc] = c[v];
      }
    }
  }
}

// ---------------------------------------------- per-token cosine sim --------
__global__ void sim_kernel(const float* __restrict__ bp, const int* __restrict__ order,
                           const float* __restrict__ vchrom, float* __restrict__ sim_raw) {
  __shared__ float red[5][256];
  const int token = blockIdx.x;           // b*Nn + n
  const int b = token / Nn;
  const int t = threadIdx.x;
  const float* crow = bp + ((size_t)b * Nn + order[token]) * D1 + 1;
  const float* vrow = vchrom + (size_t)token * Dd;
  float sc = 0.f, sc2 = 0.f, sv = 0.f, sv2 = 0.f, scv = 0.f;
  for (int d = t; d < Dd; d += 256) {
    float cx = crow[d], vx = vrow[d];
    sc += cx; sc2 += cx * cx; sv += vx; sv2 += vx * vx; scv += cx * vx;
  }
  red[0][t] = sc; red[1][t] = sc2; red[2][t] = sv; red[3][t] = sv2; red[4][t] = scv;
  __syncthreads();
  for (int o = 128; o > 0; o >>= 1) {
    if (t < o) for (int q = 0; q < 5; ++q) red[q][t] += red[q][t + o];
    __syncthreads();
  }
  if (t == 0) {
    sc = red[0][0]; sc2 = red[1][0]; sv = red[2][0]; sv2 = red[3][0]; scv = red[4][0];
    const float M = 2.f * (float)Dd;     // 4000
    float mu  = (sc + sv) / M;
    float var = (sc2 + sv2 - M * mu * mu) / (M - 1.f);
    float s   = sqrtf(var) + 1e-8f;
    float X = scv - mu * sv - mu * sc + (float)Dd * mu * mu;  // Σ(c-μ)(v-μ)
    float P = sc2 - 2.f * mu * sc + (float)Dd * mu * mu;
    float Q = sv2 - 2.f * mu * sv + (float)Dd * mu * mu;
    float fdot  = X / (s * s);
    float denom = fmaxf((sqrtf(P) / s) * (sqrtf(Q) / s), 1e-8f);
    sim_raw[token] = fdot / denom;
  }
}

__global__ void simstats_kernel(const float* __restrict__ sim_raw, float* __restrict__ stats) {
  __shared__ float s1[256], s2[256];
  const int b = blockIdx.x, t = threadIdx.x;
  float sum = 0.f, sq = 0.f;
  for (int i = t; i < Nn; i += 256) { float x = sim_raw[b * Nn + i]; sum += x; sq += x * x; }
  s1[t] = sum; s2[t] = sq; __syncthreads();
  for (int o = 128; o > 0; o >>= 1) {
    if (t < o) { s1[t] += s1[t + o]; s2[t] += s2[t + o]; }
    __syncthreads();
  }
  if (t == 0) {
    float mu  = s1[0] / (float)Nn;
    float var = (s2[0] - (float)Nn * mu * mu) / (float)(Nn - 1);
    stats[b * 2] = mu; stats[b * 2 + 1] = sqrtf(var);
  }
}

// ----------------------------------------- CRH MLP -> paramcr (sigmoid) -----
__global__ void cr_kernel(const float* __restrict__ ftoken, const float* __restrict__ sim_raw,
                          const float* __restrict__ stats,
                          const float* __restrict__ w1, const float* __restrict__ b1,
                          const float* __restrict__ lg, const float* __restrict__ lb,
                          const float* __restrict__ w2, const float* __restrict__ b2,
                          float* __restrict__ paramcr) {
  int token = blockIdx.x * blockDim.x + threadIdx.x;
  if (token >= Bb * Nn) return;
  int b = token / Nn;
  float sim = (sim_raw[token] - stats[b * 2]) / stats[b * 2 + 1];
  float f = ftoken[(size_t)token * 2], r = ftoken[(size_t)token * 2 + 1];
  float sum = 0.f, sq = 0.f;
  for (int o = 0; o < H2; ++o) {
    float a = fmaxf(f * w1[o * 3] + r * w1[o * 3 + 1] + sim * w1[o * 3 + 2] + b1[o], 0.f);
    sum += a; sq += a * a;
  }
  float mu  = sum / (float)H2;
  float inv = rsqrtf(sq / (float)H2 - mu * mu + 1e-5f);
  float acc = b2[0];
  for (int o = 0; o < H2; ++o) {
    float a = fmaxf(f * w1[o * 3] + r * w1[o * 3 + 1] + sim * w1[o * 3 + 2] + b1[o], 0.f);
    acc += ((a - mu) * inv * lg[o] + lb[o]) * w2[o];
  }
  paramcr[token] = 1.f / (1.f + expf(-acc));
}

// --------------------------------------------- gumbel offspring + select ----
__device__ __forceinline__ float off_val(float crv, float xr, float u0, float u1,
                                         float cx, float vx) {
  float g0 = -logf(-logf(u0 + 1e-10f) + 1e-10f);
  float g1 = -logf(-logf(u1 + 1e-10f) + 1e-10f);
  float l0 = crv + g0, l1 = xr + g1;
  float m = fmaxf(l0, l1);
  float e0 = expf(l0 - m), e1 = expf(l1 - m);
  float inv = 1.f / (e0 + e1);
  float s0 = e0 * inv, s1 = e1 * inv;
  float h0 = (l0 >= l1) ? 1.f : 0.f;
  float h1 = 1.f - h0;
  float y0 = (h0 + s0) - s0;   // straight-through forward, matches eval order
  float y1 = (h1 + s1) - s1;
  return y0 * cx + y1 * vx;
}

__global__ void offspring_kernel(const float* __restrict__ bp, const int* __restrict__ order,
                                 const float* __restrict__ vchrom,
                                 const float* __restrict__ fit_sorted,
                                 const float* __restrict__ paramcr,
                                 const float* __restrict__ cross_rand,
                                 const float* __restrict__ gumbel_u,
                                 float* __restrict__ nextPop) {
  __shared__ float red[256];
  __shared__ float s_fit;
  __shared__ int   s_sel;
  const int token = blockIdx.x;
  const int b = token / Nn;
  const int t = threadIdx.x;
  const float* crow = bp + ((size_t)b * Nn + order[token]) * D1 + 1;
  const float* vrow = vchrom + (size_t)token * Dd;
  const float* xr   = cross_rand + (size_t)token * Dd;
  const float* gu   = gumbel_u + (size_t)token * Dd * 2;
  const float crv = paramcr[token];
  float ss = 0.f;
  for (int d = t; d < Dd; d += 256) {
    float o = off_val(crv, xr[d], gu[2 * d], gu[2 * d + 1], crow[d], vrow[d]);
    ss += o * o;
  }
  red[t] = ss; __syncthreads();
  for (int o = 128; o > 0; o >>= 1) {
    if (t < o) red[t] += red[t + o];
    __syncthreads();
  }
  if (t == 0) {
    float fo = red[0];
    s_fit = fo;
    s_sel = (fo < fit_sorted[token]) ? 1 : 0;
  }
  __syncthreads();
  const bool sel = (s_sel != 0);
  float* outRow = nextPop + (size_t)token * D1;
  if (t == 0) outRow[0] = sel ? s_fit : fit_sorted[token];
  for (int d = t; d < Dd; d += 256) {
    float cx = crow[d];
    float o = off_val(crv, xr[d], gu[2 * d], gu[2 * d + 1], cx, vrow[d]);
    outRow[1 + d] = sel ? o : cx;
  }
}

// ---------------------------------------------------------------------------
extern "C" void kernel_launch(void* const* d_in, const int* in_sizes, int n_in,
                              void* d_out, int out_size, void* d_ws, size_t ws_size,
                              hipStream_t stream) {
  (void)in_sizes; (void)n_in; (void)out_size; (void)ws_size;
  const float* bp        = (const float*)d_in[0];
  const float* w_w       = (const float*)d_in[1];
  const float* w_b       = (const float*)d_in[2];
  const float* ln1_g     = (const float*)d_in[3];
  const float* ln1_b     = (const float*)d_in[4];
  const float* fq_w      = (const float*)d_in[5];
  const float* fq_b      = (const float*)d_in[6];
  const float* fk_w      = (const float*)d_in[7];
  const float* fk_b      = (const float*)d_in[8];
  const float* cr_w1     = (const float*)d_in[9];
  const float* cr_b1     = (const float*)d_in[10];
  const float* cr_ln_g   = (const float*)d_in[11];
  const float* cr_ln_b   = (const float*)d_in[12];
  const float* cr_w2     = (const float*)d_in[13];
  const float* cr_b2     = (const float*)d_in[14];
  const float* mut_rand  = (const float*)d_in[15];
  const float* crossr    = (const float*)d_in[16];
  const float* gum       = (const float*)d_in[17];

  float* out     = (float*)d_out;
  float* outNext = out;                                     // (B,N,D+1)
  float* outA    = out + (size_t)Bb * Nn * D1;              // (B,N,N)
  float* outCr   = outA + (size_t)Bb * Nn * Nn;             // (B,N)

  char* ws = (char*)d_ws;
  size_t off = 0;
  auto take = [&](size_t bytes) -> char* {
    char* p = ws + off;
    off = (off + bytes + 255) & ~(size_t)255;
    return p;
  };
  int*      order      = (int*)     take((size_t)Bb * Nn * sizeof(int));
  float*    fit_sorted = (float*)   take((size_t)Bb * Nn * sizeof(float));
  float*    ftoken     = (float*)   take((size_t)Bb * Nn * 2 * sizeof(float));
  float*    hbuf       = (float*)   take((size_t)Bb * Nn * H2 * sizeof(float));
  float*    sim_raw    = (float*)   take((size_t)Bb * Nn * sizeof(float));
  float*    simstats   = (float*)   take((size_t)Bb * 2 * sizeof(float));
  _Float16* q1         = (_Float16*)take((size_t)Bb * NP2 * KP * sizeof(_Float16));
  _Float16* k1         = (_Float16*)take((size_t)Bb * NP2 * KP * sizeof(_Float16));
  _Float16* Ah         = (_Float16*)take((size_t)Bb * NP2 * KP * sizeof(_Float16));
  _Float16* chromT     = (_Float16*)take((size_t)Bb * DP * KP * sizeof(_Float16));
  float*    vchrom     = (float*)   take((size_t)Bb * Nn * Dd * sizeof(float));

  sort_kernel    <<<Bb, 1024, 0, stream>>>(bp, order);
  ftoken_kernel  <<<Bb, 256, 0, stream>>>(bp, order, fit_sorted, ftoken);
  h_kernel       <<<(Bb * Nn + 255) / 256, 256, 0, stream>>>(ftoken, w_w, w_b, ln1_g, ln1_b, hbuf);
  qk_kernel      <<<Bb * NP2, 256, 0, stream>>>(hbuf, fq_w, fq_b, fk_w, fk_b, q1, k1);
  chromT_kernel  <<<(int)(((size_t)Bb * DP * KP + 255) / 256), 256, 0, stream>>>(bp, order, chromT);
  gemm1_kernel   <<<dim3(NP2 / 32, NP2 / 64, Bb), 32, 0, stream>>>(q1, k1, mut_rand, outA, Ah);
  gemm2_kernel   <<<dim3(NP2 / 32, DP / 64, Bb), 32, 0, stream>>>(Ah, chromT, vchrom);
  sim_kernel     <<<Bb * Nn, 256, 0, stream>>>(bp, order, vchrom, sim_raw);
  simstats_kernel<<<Bb, 256, 0, stream>>>(sim_raw, simstats);
  cr_kernel      <<<(Bb * Nn + 255) / 256, 256, 0, stream>>>(ftoken, sim_raw, simstats,
                                                             cr_w1, cr_b1, cr_ln_g, cr_ln_b,
                                                             cr_w2, cr_b2, outCr);
  offspring_kernel<<<Bb * Nn, 256, 0, stream>>>(bp, order, vchrom, fit_sorted, outCr,
                                                crossr, gum, outNext);
}
